// TransformerState_62380105007373
// MI455X (gfx1250) — matrix operand order repaired and checked by reading, not compiled, not run on hardware
//
#include <hip/hip_runtime.h>

// ---------------- problem constants (from reference) ----------------
#define SEQ_LEN   65536
#define NQ        256
#define DK        1040      // TYPE_DIM(16) + BYTES_DIM(1024)
#define DPAD      1056      // padded to multiple of 32 (zero fill)
#define DV        1024
#define MEMD      2064
#define TYPE_ID   4
#define MSCALE    1000000.0f

// ---------------- tiling ----------------
#define SCHUNK    128                   // keys per block
#define NCHUNK    (SEQ_LEN / SCHUNK)    // 512
#define QS        40                    // LDS row stride in halfs
#define KS        40
#define NSTEP     (DPAD / 32)           // 33 d-steps

typedef _Float16 h16 __attribute__((ext_vector_type(16)));
typedef _Float16 h8  __attribute__((ext_vector_type(8)));
typedef float    f8  __attribute__((ext_vector_type(8)));

__device__ __forceinline__ unsigned f2ord(float f) {
  unsigned u = __float_as_uint(f);
  return (u & 0x80000000u) ? ~u : (u | 0x80000000u);
}

__device__ __forceinline__ h8 cvt8(float4 a, float4 b) {
  h8 r;
  r[0] = (_Float16)a.x; r[1] = (_Float16)a.y; r[2] = (_Float16)a.z; r[3] = (_Float16)a.w;
  r[4] = (_Float16)b.x; r[5] = (_Float16)b.y; r[6] = (_Float16)b.z; r[7] = (_Float16)b.w;
  return r;
}

struct StageRegs { float4 q0, q1, k0, k1, k2, k3; };

// ================================================================
// Kernel 1: block = 64 queries x 128 keys. Per d-step (32 dims):
// stage Q(64x32)+K(128x32) f32->f16 into double-buffered LDS; each
// of 8 waves issues 4x v_wmma_f32_16x16x32_f16 (A fragment reused).
// Software pipeline: global loads -> WMMAs -> cvt/ds_store -> barrier.
// Running argmax of match*1e6+pos -> packed u64 atomicMax.
// ================================================================
__global__ void __launch_bounds__(256) ts_score_argmax(
    const float* __restrict__ tokens,
    const float* __restrict__ addrq,
    unsigned long long* __restrict__ best)
{
  __shared__ __align__(16) _Float16 QsBuf[2][64 * QS];   // 2 x 5.0 KB
  __shared__ __align__(16) _Float16 KsBuf[2][128 * KS];  // 2 x 10.0 KB

  const int tid  = threadIdx.x;
  const int lane = tid & 31;
  const int wave = tid >> 5;          // 0..7
  const int qt   = wave & 3;          // query sub-tile (16 queries)
  const int kg   = wave >> 2;         // key group (0/1) -> 64 keys each
  const int hl   = lane >> 4;
  const int lm   = lane & 15;

  const int qbase   = blockIdx.x * 64;
  const int schunk0 = blockIdx.y * SCHUNK;

  // staging thread mapping
  const int qrow = tid >> 2;            // 0..63
  const int qc0  = (tid & 3) * 8;       // 0,8,16,24
  const int krow = tid >> 1;            // 0..127
  const int kc0  = (tid & 1) * 16;      // 0,16
  // qp + d == q[qrow][d] within the addr_query region (d >= 16)
  const float* __restrict__ qp = addrq + (size_t)(qbase + qrow) * 1024 + (qc0 - 16);
  const float* __restrict__ kp = tokens + (size_t)(schunk0 + krow) * MEMD + kc0;

  const float4 FZ = make_float4(0.f, 0.f, 0.f, 0.f);

  // ---- staging: loads (to regs) / store (cvt + ds_store_b128) ----
  auto loadMid = [&](int d0) {                       // 32 <= d0 <= 992: branch-free
    StageRegs r;
    r.q0 = *(const float4*)(qp + d0);
    r.q1 = *(const float4*)(qp + d0 + 4);
    r.k0 = *(const float4*)(kp + d0);
    r.k1 = *(const float4*)(kp + d0 + 4);
    r.k2 = *(const float4*)(kp + d0 + 8);
    r.k3 = *(const float4*)(kp + d0 + 12);
    return r;
  };
  auto loadFirst = [&]() {                           // d0 = 0: type one-hot in q
    StageRegs r;
    if (qc0 < 16) {
      r.q0 = FZ;
      r.q1 = (qc0 == 0) ? make_float4(1.f, 0.f, 0.f, 0.f) : FZ;  // d=4 -> 1.0
    } else {
      r.q0 = *(const float4*)(qp + 0);
      r.q1 = *(const float4*)(qp + 4);
    }
    r.k0 = *(const float4*)(kp + 0);
    r.k1 = *(const float4*)(kp + 4);
    r.k2 = *(const float4*)(kp + 8);
    r.k3 = *(const float4*)(kp + 12);
    return r;
  };
  auto loadLast = [&]() {                            // d0 = 1024: zero pad past 1040
    StageRegs r;
    if (qc0 < 16) {
      r.q0 = *(const float4*)(qp + 1024);
      r.q1 = *(const float4*)(qp + 1028);
    } else { r.q0 = FZ; r.q1 = FZ; }
    if (kc0 == 0) {
      r.k0 = *(const float4*)(kp + 1024);
      r.k1 = *(const float4*)(kp + 1028);
      r.k2 = *(const float4*)(kp + 1032);
      r.k3 = *(const float4*)(kp + 1036);
    } else { r.k0 = FZ; r.k1 = FZ; r.k2 = FZ; r.k3 = FZ; }
    return r;
  };
  auto storeStage = [&](const StageRegs& r, _Float16* Qb, _Float16* Kb) {
    *(h8*)&Qb[qrow * QS + qc0]     = cvt8(r.q0, r.q1);
    *(h8*)&Kb[krow * KS + kc0]     = cvt8(r.k0, r.k1);
    *(h8*)&Kb[krow * KS + kc0 + 8] = cvt8(r.k2, r.k3);
  };

  f8 acc[4] = {{}, {}, {}, {}};

  auto compute = [&](const _Float16* Qb, const _Float16* Kb) {
    union { h16 v; h8 h[2]; } A, B;
    // A 16x32 f16 layout: lanes0-15 K=0..7 & 16..23; lanes16-31 K=8..15 & 24..31
    A.h[0] = *(const h8*)&Qb[(qt * 16 + lm) * QS + 8 * hl];
    A.h[1] = *(const h8*)&Qb[(qt * 16 + lm) * QS + 16 + 8 * hl];
#pragma unroll
    for (int j = 0; j < 4; ++j) {
      const int kr = kg * 64 + j * 16 + lm;
      // B 32x16 f16 layout: lanes0-15 K=0..7 & 16..23; lanes16-31 K=8..15 & 24..31
      B.h[0] = *(const h8*)&Kb[kr * KS + 8 * hl];
      B.h[1] = *(const h8*)&Kb[kr * KS + 16 + 8 * hl];
      acc[j] = __builtin_amdgcn_wmma_f32_16x16x32_f16(false, A.v, false, B.v,
                                                      (short)0, acc[j], false, false);
    }
  };

  // ---- prologue: stage step 0 (FIRST) ----
  {
    StageRegs r = loadFirst();
    storeStage(r, QsBuf[0], KsBuf[0]);
  }
  __syncthreads();

  int cur = 0;
  // ---- hot loop: steps 0..30 computed, steps 1..31 staged (all MID, no branches) ----
#pragma unroll 1
  for (int i = 0; i < NSTEP - 2; ++i) {
    StageRegs r = loadMid((i + 1) * 32);             // issue global loads
    compute(QsBuf[cur], KsBuf[cur]);                 // WMMAs while loads in flight
    storeStage(r, QsBuf[cur ^ 1], KsBuf[cur ^ 1]);   // wait/cvt/store after compute
    __syncthreads();
    cur ^= 1;
  }
  // ---- step 31 computed, LAST (d0=1024) staged ----
  {
    StageRegs r = loadLast();
    compute(QsBuf[cur], KsBuf[cur]);
    storeStage(r, QsBuf[cur ^ 1], KsBuf[cur ^ 1]);
    __syncthreads();
    cur ^= 1;
  }
  // ---- step 32 computed, nothing staged ----
  compute(QsBuf[cur], KsBuf[cur]);

  // ---- running argmax over the 4 accumulated 16x16 tiles ----
  float    bVal[8];
  unsigned bIdx[8];
#pragma unroll
  for (int r = 0; r < 8; ++r) { bVal[r] = -3.4e38f; bIdx[r] = 0u; }

#pragma unroll
  for (int j = 0; j < 4; ++j) {
    const unsigned skey  = (unsigned)(schunk0 + kg * 64 + j * 16 + lm);
    const float    skeyf = (float)skey;
#pragma unroll
    for (int r = 0; r < 8; ++r) {
      const float comb = acc[j][r] * MSCALE + skeyf;  // same fp32 expr as reference
      if (comb > bVal[r]) { bVal[r] = comb; bIdx[r] = skey; }
    }
  }

  // reduce across the 16 key lanes inside each half-wave
#pragma unroll
  for (int m = 1; m <= 8; m <<= 1) {
#pragma unroll
    for (int r = 0; r < 8; ++r) {
      const float    ov = __shfl_xor(bVal[r], m, 32);
      const unsigned oi = (unsigned)__shfl_xor((int)bIdx[r], m, 32);
      if (ov > bVal[r] || (ov == bVal[r] && oi > bIdx[r])) { bVal[r] = ov; bIdx[r] = oi; }
    }
  }

  if (lm == 0) {
#pragma unroll
    for (int r = 0; r < 8; ++r) {
      const int q = qbase + qt * 16 + r + 8 * hl;   // C layout: VGPR r -> query r+8*hl
      const unsigned long long packed =
          ((unsigned long long)f2ord(bVal[r]) << 32) | (unsigned long long)bIdx[r];
      atomicMax(&best[q], packed);
    }
  }
}

// ================================================================
// Kernel 0: zero the argmax slots (harness poisons d_ws)
// ================================================================
__global__ void ts_init(unsigned long long* __restrict__ best) {
  best[threadIdx.x] = 0ull;
}

// ================================================================
// Kernel 2: out[b] = v[argmax_b] = tokens[idx, 1040:2064], exact fp32
// ================================================================
__global__ void __launch_bounds__(256) ts_gather(
    const float* __restrict__ tokens,
    const unsigned long long* __restrict__ best,
    float* __restrict__ out)
{
  const int b = blockIdx.x;
  const unsigned idx = (unsigned)(best[b] & 0xFFFFFFFFull);
  const float4* __restrict__ v = (const float4*)(tokens + (size_t)idx * MEMD + DK);
  float4* __restrict__ o = (float4*)(out + (size_t)b * DV);
  o[threadIdx.x] = v[threadIdx.x];   // 256 threads x float4 = 1024 floats
}

extern "C" void kernel_launch(void* const* d_in, const int* in_sizes, int n_in,
                              void* d_out, int out_size, void* d_ws, size_t ws_size,
                              hipStream_t stream) {
  const float* tokens = (const float*)d_in[0];   // (65536, 2064) f32
  const float* addrq  = (const float*)d_in[1];   // (256, 1024)  f32
  float* out = (float*)d_out;                    // (256, 4, 256) f32 flat
  unsigned long long* best = (unsigned long long*)d_ws;  // 256 * 8B

  ts_init<<<1, NQ, 0, stream>>>(best);
  dim3 grid(4, NCHUNK);   // x fastest -> the 4 blocks sharing a key chunk co-schedule (L2 dedup)
  ts_score_argmax<<<grid, 256, 0, stream>>>(tokens, addrq, best);
  ts_gather<<<NQ, 256, 0, stream>>>(tokens, best, out);
}
